// Seq2SeqLSTM_34454227648929
// MI455X (gfx1250) — matrix operand order repaired and checked by reading, not compiled, not run on hardware
//
#include <hip/hip_runtime.h>
#include <hip/hip_bf16.h>

// Problem constants (from setup_inputs): B=256, S=512, F=64, H=1024, T=64
#define B_   256
#define S_   512
#define F_   64
#define H_   1024
#define T_   64
#define KCAT (F_ + H_)          // 1088 : concatenated K = [x | h]
#define KCH  (KCAT / 32)        // 34 k-chunks of 32

typedef unsigned short u16;
typedef __attribute__((ext_vector_type(16))) __bf16          v16bf;
typedef __attribute__((ext_vector_type(8)))  float           v8f;
typedef __attribute__((ext_vector_type(8)))  unsigned short  ushx8;

union FragU { ushx8 u[2]; v16bf v; };

__device__ __forceinline__ u16 f2bf(float x) {
  unsigned int u = __float_as_uint(x);
  u += 0x7FFFu + ((u >> 16) & 1u);      // round-to-nearest-even
  return (u16)(u >> 16);
}
__device__ __forceinline__ float sig_(float x) {
  x = fminf(fmaxf(x, -30.f), 30.f);
  return 1.0f / (1.0f + __expf(-x));
}
__device__ __forceinline__ float tanh_(float x) {
  x = fminf(fmaxf(x, -15.f), 15.f);
  float e = __expf(2.0f * x);
  return (e - 1.0f) / (e + 1.0f);
}

// A fragment (16x32 bf16): lane holds row M=lane&15, 16 consecutive K at koff=(lane>>4)*16
__device__ __forceinline__ v16bf load_a(const u16* arow, int koff) {
  FragU t;
  t.u[0] = *(const ushx8*)(arow + koff);
  t.u[1] = *(const ushx8*)(arow + koff + 8);
  return t.v;
}
// B fragment (32x16 bf16): lane holds col N=lane&15, 16 consecutive K at kb=(lane>>4)*16
// Streamed-once panel -> nontemporal so it doesn't thrash the WGP$.
__device__ __forceinline__ v16bf load_b(const u16* bp) {
  FragU t;
  t.u[0] = __builtin_nontemporal_load((const ushx8*)(bp));
  t.u[1] = __builtin_nontemporal_load((const ushx8*)(bp + 8));
  return t.v;
}

// ---------------------------------------------------------------------------
// Fused LSTM step:  gates = [x|h] @ Wcat^T + b  ->  (c,h) update
// grid (8, 16), block 256.  Block = 32 batch rows x 64 hidden cols (all 4 gates).
// Wave: m_sub = w&1 (16 rows), n_sub = w>>1 (16 cols) across the 4 gates.
// Software-pipelined: prefetch chunk kc+1 while WMMA'ing chunk kc.
// ---------------------------------------------------------------------------
__global__ __launch_bounds__(256)
void lstm_step(const u16* __restrict__ xbase, int xstride,
               const u16* __restrict__ h_in,
               const u16* __restrict__ Wcat,
               const float* __restrict__ bih, const float* __restrict__ bhh,
               float* __restrict__ cst,
               u16* __restrict__ h_out)
{
  const int lane  = threadIdx.x & 31;
  const int w     = threadIdx.x >> 5;
  const int m_sub = w & 1;       // 2 row sub-tiles of 16
  const int n_sub = w >> 1;      // 4 col sub-tiles of 16
  const int m0 = blockIdx.x * 32;
  const int n0 = blockIdx.y * 64;

  const int am   = m0 + 16 * m_sub + (lane & 15);
  const int koff = (lane >> 4) * 16;
  const int kb   = (lane >> 4) * 16;

  const v8f vzero = {};
  v8f acc[4];
#pragma unroll
  for (int g = 0; g < 4; ++g) acc[g] = vzero;

  // Per-gate B row pointers; chunk kc lives at +kc*32 (Wcat K is already [x|h]).
  const u16* bp[4];
  const int hcol = n0 + n_sub * 16 + (lane & 15);
#pragma unroll
  for (int g = 0; g < 4; ++g)
    bp[g] = Wcat + (size_t)(g * H_ + hcol) * KCAT + kb;

  // A row bases: k<64 from x, k>=64 from h (indexed directly by k).
  const u16* arow_x = xbase + (size_t)am * xstride;
  const u16* arow_h = h_in + (size_t)am * H_ - F_;

  // ---- pipelined K loop ----
  v16bf a_cur = load_a(arow_x, koff);
  v16bf b_cur0 = load_b(bp[0]);
  v16bf b_cur1 = load_b(bp[1]);
  v16bf b_cur2 = load_b(bp[2]);
  v16bf b_cur3 = load_b(bp[3]);

#pragma unroll 2
  for (int kc = 0; kc < KCH - 1; ++kc) {
    const int kn = (kc + 1) * 32;
    const u16* an = (kn < F_) ? (arow_x + kn) : (arow_h + kn);
    v16bf a_nxt  = load_a(an, koff);
    v16bf b_nxt0 = load_b(bp[0] + kn);
    v16bf b_nxt1 = load_b(bp[1] + kn);
    v16bf b_nxt2 = load_b(bp[2] + kn);
    v16bf b_nxt3 = load_b(bp[3] + kn);

    acc[0] = __builtin_amdgcn_wmma_f32_16x16x32_bf16(false, a_cur, false, b_cur0, (short)0, acc[0], false, false);
    acc[1] = __builtin_amdgcn_wmma_f32_16x16x32_bf16(false, a_cur, false, b_cur1, (short)0, acc[1], false, false);
    acc[2] = __builtin_amdgcn_wmma_f32_16x16x32_bf16(false, a_cur, false, b_cur2, (short)0, acc[2], false, false);
    acc[3] = __builtin_amdgcn_wmma_f32_16x16x32_bf16(false, a_cur, false, b_cur3, (short)0, acc[3], false, false);

    a_cur = a_nxt;
    b_cur0 = b_nxt0; b_cur1 = b_nxt1; b_cur2 = b_nxt2; b_cur3 = b_nxt3;
  }
  acc[0] = __builtin_amdgcn_wmma_f32_16x16x32_bf16(false, a_cur, false, b_cur0, (short)0, acc[0], false, false);
  acc[1] = __builtin_amdgcn_wmma_f32_16x16x32_bf16(false, a_cur, false, b_cur1, (short)0, acc[1], false, false);
  acc[2] = __builtin_amdgcn_wmma_f32_16x16x32_bf16(false, a_cur, false, b_cur2, (short)0, acc[2], false, false);
  acc[3] = __builtin_amdgcn_wmma_f32_16x16x32_bf16(false, a_cur, false, b_cur3, (short)0, acc[3], false, false);

  // Elementwise LSTM cell update (PyTorch gate order i,f,g,o)
  const int nh = hcol;
  float bi = bih[0 * H_ + nh] + bhh[0 * H_ + nh];
  float bf = bih[1 * H_ + nh] + bhh[1 * H_ + nh];
  float bg = bih[2 * H_ + nh] + bhh[2 * H_ + nh];
  float bo = bih[3 * H_ + nh] + bhh[3 * H_ + nh];
#pragma unroll
  for (int r = 0; r < 8; ++r) {
    int m = m0 + 16 * m_sub + r + ((lane >> 4) << 3);   // WMMA C layout row
    size_t idx = (size_t)m * H_ + nh;
    float gi = sig_(acc[0][r] + bi);
    float gf = sig_(acc[1][r] + bf);
    float gg = tanh_(acc[2][r] + bg);
    float go = sig_(acc[3][r] + bo);
    float cn = gf * cst[idx] + gi * gg;
    cst[idx] = cn;
    h_out[idx] = f2bf(go * tanh_(cn));
  }
}

// ---------------------------------------------------------------------------
// Decoder projection: pred = h @ W_fc^T + b_fc ; write f32 output slice + bf16 x
// grid (4), block 256. Pipelined like lstm_step.
// ---------------------------------------------------------------------------
__global__ __launch_bounds__(256)
void pred_step(const u16* __restrict__ h, const u16* __restrict__ Wfc,
               const float* __restrict__ bfc,
               float* __restrict__ out, u16* __restrict__ xbf, int t)
{
  const int lane   = threadIdx.x & 31;
  const int w      = threadIdx.x >> 5;
  const int m_sub  = w & 3;
  const int n_pair = w >> 2;
  const int m0 = blockIdx.x * 64;
  const int am   = m0 + 16 * m_sub + (lane & 15);
  const int koff = (lane >> 4) * 16;
  const int kb   = (lane >> 4) * 16;

  const v8f vzero = {};
  v8f acc0 = vzero, acc1 = vzero;
  const u16* bp0 = Wfc + (size_t)((n_pair * 2 + 0) * 16 + (lane & 15)) * H_ + kb;
  const u16* bp1 = Wfc + (size_t)((n_pair * 2 + 1) * 16 + (lane & 15)) * H_ + kb;

  const u16* arow = h + (size_t)am * H_;

  v16bf a_cur  = load_a(arow, koff);
  v16bf b_cur0 = load_b(bp0);
  v16bf b_cur1 = load_b(bp1);
#pragma unroll 2
  for (int kc = 0; kc < H_ / 32 - 1; ++kc) {
    const int kn = (kc + 1) * 32;
    v16bf a_nxt  = load_a(arow + kn, koff);
    v16bf b_nxt0 = load_b(bp0 + kn);
    v16bf b_nxt1 = load_b(bp1 + kn);
    acc0 = __builtin_amdgcn_wmma_f32_16x16x32_bf16(false, a_cur, false, b_cur0, (short)0, acc0, false, false);
    acc1 = __builtin_amdgcn_wmma_f32_16x16x32_bf16(false, a_cur, false, b_cur1, (short)0, acc1, false, false);
    a_cur = a_nxt; b_cur0 = b_nxt0; b_cur1 = b_nxt1;
  }
  acc0 = __builtin_amdgcn_wmma_f32_16x16x32_bf16(false, a_cur, false, b_cur0, (short)0, acc0, false, false);
  acc1 = __builtin_amdgcn_wmma_f32_16x16x32_bf16(false, a_cur, false, b_cur1, (short)0, acc1, false, false);

#pragma unroll
  for (int ns = 0; ns < 2; ++ns) {
    int n = (n_pair * 2 + ns) * 16 + (lane & 15);
    float bb = bfc[n];
    const v8f& a = ns ? acc1 : acc0;
#pragma unroll
    for (int r = 0; r < 8; ++r) {
      int m = m0 + 16 * m_sub + r + ((lane >> 4) << 3);
      float v = a[r] + bb;
      out[(size_t)m * (T_ * F_) + (size_t)t * F_ + n] = v;   // (B,T,F)
      xbf[m * F_ + n] = f2bf(v);
    }
  }
}

// ---------------------------------------------------------------------------
// Prep kernels
// ---------------------------------------------------------------------------
__global__ void pack_wcat(const float* __restrict__ Wih, const float* __restrict__ Whh,
                          u16* __restrict__ Wcat, int total)
{
  int i = blockIdx.x * blockDim.x + threadIdx.x;
  if (i >= total) return;
  int row = i / KCAT, col = i % KCAT;
  float v = (col < F_) ? Wih[row * F_ + col]
                       : Whh[(size_t)row * H_ + (col - F_)];
  Wcat[i] = f2bf(v);
}

__global__ void cvt_bf(const float* __restrict__ src, u16* __restrict__ dst, int n)
{
  int i = blockIdx.x * blockDim.x + threadIdx.x;
  if (i < n) dst[i] = f2bf(src[i]);
}

// ---------------------------------------------------------------------------
// Host launcher
// ---------------------------------------------------------------------------
extern "C" void kernel_launch(void* const* d_in, const int* in_sizes, int n_in,
                              void* d_out, int out_size, void* d_ws, size_t ws_size,
                              hipStream_t stream)
{
  const float* x_seq = (const float*)d_in[0];
  const float* Wih_e = (const float*)d_in[1];
  const float* Whh_e = (const float*)d_in[2];
  const float* bih_e = (const float*)d_in[3];
  const float* bhh_e = (const float*)d_in[4];
  const float* Wih_d = (const float*)d_in[5];
  const float* Whh_d = (const float*)d_in[6];
  const float* bih_d = (const float*)d_in[7];
  const float* bhh_d = (const float*)d_in[8];
  const float* Wfc   = (const float*)d_in[9];
  const float* bfc   = (const float*)d_in[10];
  float* out = (float*)d_out;

  // Workspace carve-up (all chunk sizes are 256B multiples); ~35.2 MB total
  char* ws = (char*)d_ws;
  u16* Wcat_e  = (u16*)ws;  ws += (size_t)4 * H_ * KCAT * 2;  // 8.9 MB
  u16* Wcat_d  = (u16*)ws;  ws += (size_t)4 * H_ * KCAT * 2;  // 8.9 MB
  u16* Wfc_bf  = (u16*)ws;  ws += (size_t)F_ * H_ * 2;        // 128 KB
  u16* xseq_bf = (u16*)ws;  ws += (size_t)B_ * S_ * F_ * 2;   // 16.8 MB
  u16* h0      = (u16*)ws;  ws += (size_t)B_ * H_ * 2;        // 512 KB
  u16* h1      = (u16*)ws;  ws += (size_t)B_ * H_ * 2;        // 512 KB
  float* cst   = (float*)ws; ws += (size_t)B_ * H_ * 4;       // 1 MB
  u16* xbf     = (u16*)ws;  ws += (size_t)B_ * F_ * 2;        // 32 KB
  (void)ws_size; (void)in_sizes; (void)n_in; (void)out_size;

  // --- prep: pack bf16 weights, convert input sequence, zero state ---
  {
    int totW = 4 * H_ * KCAT;
    pack_wcat<<<(totW + 255) / 256, 256, 0, stream>>>(Wih_e, Whh_e, Wcat_e, totW);
    pack_wcat<<<(totW + 255) / 256, 256, 0, stream>>>(Wih_d, Whh_d, Wcat_d, totW);
    int nfc = F_ * H_;
    cvt_bf<<<(nfc + 255) / 256, 256, 0, stream>>>(Wfc, Wfc_bf, nfc);
    int nx = B_ * S_ * F_;
    cvt_bf<<<(nx + 255) / 256, 256, 0, stream>>>(x_seq, xseq_bf, nx);
    hipMemsetAsync(h0, 0, (size_t)B_ * H_ * 2, stream);   // bf16 zero == 0x0000
    hipMemsetAsync(cst, 0, (size_t)B_ * H_ * 4, stream);
  }

  dim3 grid(B_ / 32, H_ / 64), blk(256);
  u16* hbuf[2] = {h0, h1};

  // --- encoder: 512 recurrent steps (ping-pong h) ---
  for (int t = 0; t < S_; ++t) {
    lstm_step<<<grid, blk, 0, stream>>>(
        xseq_bf + (size_t)t * F_, S_ * F_,
        hbuf[t & 1], Wcat_e, bih_e, bhh_e, cst, hbuf[(t + 1) & 1]);
  }

  // --- decoder: 64 autoregressive steps ---
  for (int d = 0; d < T_; ++d) {
    const u16* xb = (d == 0) ? (xseq_bf + (size_t)(S_ - 1) * F_) : xbf;
    int xs = (d == 0) ? (S_ * F_) : F_;
    lstm_step<<<grid, blk, 0, stream>>>(
        xb, xs, hbuf[d & 1], Wcat_d, bih_d, bhh_d, cst, hbuf[(d + 1) & 1]);
    pred_step<<<dim3(B_ / 64), blk, 0, stream>>>(
        hbuf[(d + 1) & 1], Wfc_bf, bfc, out, xbf, d);
  }
}
